// GraphTripleConv_38070590112331
// MI455X (gfx1250) — compile-verified, matrix operand order checked by reading, not run on hardware
//
#include <hip/hip_runtime.h>

#define HIDDEN   256
#define PRED_OUT 128
#define B_  32
#define O_  512
#define T_  2048
#define D_  128
#define IN1 384          // 3*D
#define OUT1 640         // 2*HIDDEN + PRED_OUT

typedef __attribute__((ext_vector_type(16))) __bf16 v16bf;
typedef __attribute__((ext_vector_type(8)))  __bf16 v8bf;
typedef __attribute__((ext_vector_type(8)))  float  v8f;

__device__ __forceinline__ __bf16 f2bf(float f) {
  union { float f; unsigned u; } x; x.f = f;
  unsigned r = x.u + 0x7FFFu + ((x.u >> 16) & 1u);   // round-to-nearest-even
  unsigned short h = (unsigned short)(r >> 16);
  return __builtin_bit_cast(__bf16, h);
}

// A-fragment (16x32 bf16) per ISA layout: lane = row (M = lane&15),
// lane[4] selects K-half; VGPR0-3 = K {base..base+7}, VGPR4-7 = K {base+16..base+23}
__device__ __forceinline__ v16bf load_a_frag(const __bf16* lds, int lda,
                                             int row, int k0, int kh) {
  const __bf16* p = lds + row * lda + k0 + kh * 8;
  union { v16bf v; v8bf h[2]; } u;
  u.h[0] = *(const v8bf*)(p);
  u.h[1] = *(const v8bf*)(p + 16);
  return u.v;
}

__global__ void zero_f32(float* __restrict__ p, int n) {
  int i = blockIdx.x * blockDim.x + threadIdx.x;
  if (i < n) p[i] = 0.f;
}

// Repack W (KxN, f32 row-major) into per-lane WMMA B-fragment order (bf16).
// dst element i = ((nt*KC + kc)*32 + lane)*16 + e, with
//  n = nt*16 + (lane&15),  k = kc*32 + (e>>3)*16 + (lane>>4)*8 + (e&7)
__global__ void pack_weights(const float* __restrict__ W, __bf16* __restrict__ dst,
                             int K, int N) {
  int i = blockIdx.x * blockDim.x + threadIdx.x;
  if (i >= K * N) return;
  int e    = i & 15;
  int lane = (i >> 4) & 31;
  int tile = i >> 9;
  int KC = K >> 5;
  int kc = tile % KC;
  int nt = tile / KC;
  int n = nt * 16 + (lane & 15);
  int k = kc * 32 + ((e >> 3) * 16) + ((lane >> 4) * 8) + (e & 7);
  dst[i] = f2bf(W[(size_t)k * N + n]);
}

// ---------------- edge MLP: gather -> (384->256->640) -> scatter ----------------
__global__ __launch_bounds__(256) void edge_mlp(
    const float* __restrict__ obj, const float* __restrict__ pred,
    const int* __restrict__ edges, const unsigned char* __restrict__ ind,
    const __bf16* __restrict__ pkW1, const float* __restrict__ b1,
    const __bf16* __restrict__ pkW2, const float* __restrict__ b2,
    float* __restrict__ pooled, float* __restrict__ counts,
    float* __restrict__ out_p)
{
  constexpr int MT = 64, LDX = IN1 + 8, LDH = HIDDEN + 8;
  __shared__ __bf16 ldsX[MT * LDX];
  __shared__ __bf16 ldsH[MT * LDH];
  __shared__ int   ldsS[MT], ldsO[MT];
  __shared__ float ldsM[MT];

  const int tid = threadIdx.x;
  const int g0  = blockIdx.x * MT;       // first global edge row
  const int bb  = g0 / T_;               // batch (T_ % MT == 0)

  if (tid < MT) {
    int g = g0 + tid;
    int s = edges[2 * g + 0];
    int o = edges[2 * g + 1];
    float m = ind[g] ? 1.0f : 0.0f;
    ldsS[tid] = s; ldsO[tid] = o; ldsM[tid] = m;
    if (m != 0.0f) {
      atomicAdd(&counts[bb * O_ + s], 1.0f);
      atomicAdd(&counts[bb * O_ + o], 1.0f);
    }
  }
  __syncthreads();

  { // gather cur_t = [obj[s] | pred | obj[o]] as bf16, 4 threads per row
    int row = tid >> 2, q = tid & 3;
    int g = g0 + row;
    const float* srcS = obj  + ((size_t)bb * O_ + ldsS[row]) * D_;
    const float* srcO = obj  + ((size_t)bb * O_ + ldsO[row]) * D_;
    const float* srcP = pred + (size_t)g * D_;
    __bf16* dst = ldsX + row * LDX;
    for (int j = 0; j < 96; j += 4) {
      int col = q * 96 + j;
      const float* s4 = (col < 128) ? (srcS + col)
                      : (col < 256) ? (srcP + col - 128)
                                    : (srcO + col - 256);
      float4 v = *(const float4*)s4;
      dst[col + 0] = f2bf(v.x); dst[col + 1] = f2bf(v.y);
      dst[col + 2] = f2bf(v.z); dst[col + 3] = f2bf(v.w);
    }
  }
  __syncthreads();

  const int wv = tid >> 5, lane = tid & 31;
  const int c = lane & 15, kh = lane >> 4;
  const int rowbase = (wv >> 1) * 16;    // 4 row sub-tiles, 2 waves each
  const int par = wv & 1;                // split N between wave pair

  // GEMM1: (64x384) x (384x256), relu -> ldsH bf16
  for (int nti = 0; nti < 8; ++nti) {
    int nt = par + 2 * nti;              // 0..15
    v8f acc = {};
    #pragma unroll
    for (int kc = 0; kc < 12; ++kc) {
      v16bf a = load_a_frag(ldsX, LDX, rowbase + c, kc * 32, kh);
      v16bf b = *(const v16bf*)(pkW1 + ((size_t)(nt * 12 + kc) * 32 + lane) * 16);
      acc = __builtin_amdgcn_wmma_f32_16x16x32_bf16(false, a, false, b,
                                                    (short)0, acc, false, false);
    }
    float bias = b1[nt * 16 + c];
    #pragma unroll
    for (int r = 0; r < 8; ++r) {
      float h = acc[r] + bias;
      h = h > 0.f ? h : 0.f;
      ldsH[(rowbase + r + 8 * kh) * LDH + nt * 16 + c] = f2bf(h);
    }
  }
  __syncthreads();

  // GEMM2: (64x256) x (256x640), relu, split into scatter(new_s), out(new_p), scatter(new_o)
  for (int nti = 0; nti < 20; ++nti) {
    int nt = par + 2 * nti;              // 0..39
    v8f acc = {};
    #pragma unroll
    for (int kc = 0; kc < 8; ++kc) {
      v16bf a = load_a_frag(ldsH, LDH, rowbase + c, kc * 32, kh);
      v16bf b = *(const v16bf*)(pkW2 + ((size_t)(nt * 8 + kc) * 32 + lane) * 16);
      acc = __builtin_amdgcn_wmma_f32_16x16x32_bf16(false, a, false, b,
                                                    (short)0, acc, false, false);
    }
    int col = nt * 16 + c;               // 0..639
    float bias = b2[col];
    #pragma unroll
    for (int r = 0; r < 8; ++r) {
      int lrow = rowbase + r + 8 * kh;
      int g = g0 + lrow;
      float v = acc[r] + bias;
      v = v > 0.f ? v : 0.f;
      if (col < HIDDEN) {                                 // new_s
        if (ldsM[lrow] != 0.f)
          atomicAdd(&pooled[((size_t)bb * O_ + ldsS[lrow]) * HIDDEN + col], v);
      } else if (col < HIDDEN + PRED_OUT) {               // new_p
        out_p[(size_t)g * PRED_OUT + (col - HIDDEN)] = v;
      } else {                                            // new_o
        if (ldsM[lrow] != 0.f)
          atomicAdd(&pooled[((size_t)bb * O_ + ldsO[lrow]) * HIDDEN
                            + (col - HIDDEN - PRED_OUT)], v);
      }
    }
  }
}

// ---------------- object MLP: normalize -> (256->256->128) ----------------
__global__ __launch_bounds__(256) void obj_mlp(
    const float* __restrict__ pooled, const float* __restrict__ counts,
    const __bf16* __restrict__ pkW1, const float* __restrict__ b1,
    const __bf16* __restrict__ pkW2, const float* __restrict__ b2,
    float* __restrict__ out_obj)
{
  constexpr int MT = 64, LDX = HIDDEN + 8, LDH = HIDDEN + 8;
  __shared__ __bf16 ldsX[MT * LDX];
  __shared__ __bf16 ldsH[MT * LDH];

  const int tid = threadIdx.x;
  const int g0  = blockIdx.x * MT;       // first global object row (b*O + o)

  { // normalize by max(count,1) and convert to bf16
    int row = tid >> 2, q = tid & 3;
    int g = g0 + row;
    float inv = 1.0f / fmaxf(counts[g], 1.0f);
    const float* src = pooled + (size_t)g * HIDDEN;
    __bf16* dst = ldsX + row * LDX;
    for (int j = 0; j < 64; j += 4) {
      int col = q * 64 + j;
      float4 v = *(const float4*)(src + col);
      dst[col + 0] = f2bf(v.x * inv); dst[col + 1] = f2bf(v.y * inv);
      dst[col + 2] = f2bf(v.z * inv); dst[col + 3] = f2bf(v.w * inv);
    }
  }
  __syncthreads();

  const int wv = tid >> 5, lane = tid & 31;
  const int c = lane & 15, kh = lane >> 4;
  const int rowbase = (wv >> 1) * 16;
  const int par = wv & 1;

  // GEMM1: (64x256) x (256x256), relu -> ldsH
  for (int nti = 0; nti < 8; ++nti) {
    int nt = par + 2 * nti;              // 0..15
    v8f acc = {};
    #pragma unroll
    for (int kc = 0; kc < 8; ++kc) {
      v16bf a = load_a_frag(ldsX, LDX, rowbase + c, kc * 32, kh);
      v16bf b = *(const v16bf*)(pkW1 + ((size_t)(nt * 8 + kc) * 32 + lane) * 16);
      acc = __builtin_amdgcn_wmma_f32_16x16x32_bf16(false, a, false, b,
                                                    (short)0, acc, false, false);
    }
    float bias = b1[nt * 16 + c];
    #pragma unroll
    for (int r = 0; r < 8; ++r) {
      float h = fmaxf(acc[r] + bias, 0.f);
      ldsH[(rowbase + r + 8 * kh) * LDH + nt * 16 + c] = f2bf(h);
    }
  }
  __syncthreads();

  // GEMM2: (64x256) x (256x128), relu -> out
  for (int nti = 0; nti < 4; ++nti) {
    int nt = par + 2 * nti;              // 0..7
    v8f acc = {};
    #pragma unroll
    for (int kc = 0; kc < 8; ++kc) {
      v16bf a = load_a_frag(ldsH, LDH, rowbase + c, kc * 32, kh);
      v16bf b = *(const v16bf*)(pkW2 + ((size_t)(nt * 8 + kc) * 32 + lane) * 16);
      acc = __builtin_amdgcn_wmma_f32_16x16x32_bf16(false, a, false, b,
                                                    (short)0, acc, false, false);
    }
    int col = nt * 16 + c;
    float bias = b2[col];
    #pragma unroll
    for (int r = 0; r < 8; ++r) {
      int g = g0 + rowbase + r + 8 * kh;
      out_obj[(size_t)g * D_ + col] = fmaxf(acc[r] + bias, 0.f);
    }
  }
}

extern "C" void kernel_launch(void* const* d_in, const int* in_sizes, int n_in,
                              void* d_out, int out_size, void* d_ws, size_t ws_size,
                              hipStream_t stream) {
  const float* obj   = (const float*)d_in[0];
  const float* pred  = (const float*)d_in[1];
  const int*   edges = (const int*)d_in[2];
  const unsigned char* ind = (const unsigned char*)d_in[3];
  const float* W1a = (const float*)d_in[4];
  const float* b1a = (const float*)d_in[5];
  const float* W2a = (const float*)d_in[6];
  const float* b2a = (const float*)d_in[7];
  const float* W1b = (const float*)d_in[8];
  const float* b1b = (const float*)d_in[9];
  const float* W2b = (const float*)d_in[10];
  const float* b2b = (const float*)d_in[11];

  float*  pooled = (float*)d_ws;                                   // B*O*HIDDEN
  float*  counts = pooled + (size_t)B_ * O_ * HIDDEN;              // B*O
  __bf16* pkW1a  = (__bf16*)(counts + B_ * O_);
  __bf16* pkW2a  = pkW1a + IN1 * HIDDEN;
  __bf16* pkW1b  = pkW2a + HIDDEN * OUT1;
  __bf16* pkW2b  = pkW1b + HIDDEN * HIDDEN;

  float* out_obj = (float*)d_out;
  float* out_p   = out_obj + (size_t)B_ * O_ * D_;

  int nz = B_ * O_ * HIDDEN + B_ * O_;
  zero_f32<<<(nz + 255) / 256, 256, 0, stream>>>(pooled, nz);

  pack_weights<<<(IN1 * HIDDEN + 255) / 256, 256, 0, stream>>>(W1a, pkW1a, IN1, HIDDEN);
  pack_weights<<<(HIDDEN * OUT1 + 255) / 256, 256, 0, stream>>>(W2a, pkW2a, HIDDEN, OUT1);
  pack_weights<<<(HIDDEN * HIDDEN + 255) / 256, 256, 0, stream>>>(W1b, pkW1b, HIDDEN, HIDDEN);
  pack_weights<<<(HIDDEN * D_ + 255) / 256, 256, 0, stream>>>(W2b, pkW2b, HIDDEN, D_);

  edge_mlp<<<(B_ * T_) / 64, 256, 0, stream>>>(obj, pred, edges, ind,
                                               pkW1a, b1a, pkW2a, b2a,
                                               pooled, counts, out_p);
  obj_mlp<<<(B_ * O_) / 64, 256, 0, stream>>>(pooled, counts,
                                              pkW1b, b1b, pkW2b, b2b, out_obj);
}
